// ODETransitionFunction_3281355014428
// MI455X (gfx1250) — matrix operand order repaired and checked by reading, not compiled, not run on hardware
//
#include <hip/hip_runtime.h>

typedef __attribute__((ext_vector_type(16))) _Float16 v16h;
typedef __attribute__((ext_vector_type(8)))  float    v8f;

#define S_DIM    16
#define B_DIM    1024
#define K_DIM    128
#define NBLOCK   8
#define H_DIM    512
#define TGRID    (NBLOCK + 1)
#define HSTEP    0.5f

#define WAVES    8
#define NTHREADS (WAVES * 32)
#define ROWS     (S_DIM * B_DIM)          // 16384
#define NWG      (ROWS / (WAVES * 16))    // 128

__device__ __forceinline__ float fast_tanh(float x) {
  // tanh(x) = 1 - 2/(exp(2x)+1), exp via exp2 (v_exp_f32) + v_rcp_f32
  float e = __builtin_amdgcn_exp2f(x * 2.8853900817779268f); // 2*log2(e)
  return 1.0f - 2.0f * __builtin_amdgcn_rcpf(e + 1.0f);
}

__global__ __launch_bounds__(NTHREADS) void ode_rk4_wmma(
    const float* __restrict__ s_in, const float* __restrict__ t_in,
    const float* __restrict__ W1, const float* __restrict__ b1,
    const float* __restrict__ W2, const float* __restrict__ b2,
    float* __restrict__ out) {
  // Weights in B-fragment order: frag = 512 halves; lane holds 16 contiguous
  // halves at frag*512 + lane*16  (32B -> ds_load_b128 pair).
  __shared__ alignas(32) _Float16 w1f[4 * 32 * 512];   // 128 KB (K=128 x H=512)
  __shared__ alignas(32) _Float16 w2f[16 * 8 * 512];   // 128 KB (K=512 x N=128)
  __shared__ alignas(32) _Float16 stage[WAVES * 2048]; // 32 KB, per-wave A staging
  __shared__ float b1s[H_DIM];
  __shared__ float b2s[K_DIM];

  const int tid = threadIdx.x;

  // ---- cooperative f32 -> f16 weight load into fragment order ----
  for (int idx = tid; idx < H_DIM * K_DIM; idx += NTHREADS) {
    int n = idx >> 7, k = idx & 127;                 // W1[n][k], B elem (k,n)
    int frag = ((k >> 5) << 5) | (n >> 4);           // k0idx*32 + n0idx
    int ln   = (n & 15) | ((k & 8) << 1);            // +16 lanes for k bit3 set
    w1f[frag * 512 + ln * 16 + ((k & 7) | ((k & 16) >> 1))] = (_Float16)W1[idx];
  }
  for (int idx = tid; idx < K_DIM * H_DIM; idx += NTHREADS) {
    int j = idx >> 9, n = idx & 511;                 // W2[j][n], B elem (n,j)
    int frag = ((n >> 5) << 3) | (j >> 4);           // nchunk*8 + j0idx
    int ln   = (j & 15) | ((n & 8) << 1);
    w2f[frag * 512 + ln * 16 + ((n & 7) | ((n & 16) >> 1))] = (_Float16)W2[idx];
  }
  for (int idx = tid; idx < H_DIM; idx += NTHREADS) b1s[idx] = b1[idx];
  if (tid < K_DIM) b2s[tid] = b2[tid];
  __syncthreads();

  const int lane = tid & 31;
  const int wave = tid >> 5;
  const int col  = lane & 15;            // column inside 16-wide tile (C layout)
  const int hi8  = (lane >> 4) * 8;      // row offset for upper lane half
  const int rowbase = (blockIdx.x * WAVES + wave) * 16;
  _Float16* xs = &stage[wave * 2048];
  // A-fragment scatter pieces for element (m = r+hi8, k = kt*16+col):
  //   addr = (kt>>1)*512 + m*16 + 256*bit3(col) + (col&7) + 8*(kt&1)
  const int aoff = ((col >> 3) & 1) * 256 + (col & 7);

  // ---- load persistent state x into C-fragment registers ----
  float xv[64];
#pragma unroll
  for (int kt = 0; kt < 8; ++kt)
#pragma unroll
    for (int r = 0; r < 8; ++r)
      xv[kt * 8 + r] = s_in[(rowbase + r + hi8) * K_DIM + kt * 16 + col];

  float b2v[8];
#pragma unroll
  for (int j = 0; j < 8; ++j) b2v[j] = b2s[j * 16 + col];

#pragma clang loop unroll(disable)
  for (int blk = 0; blk < NBLOCK; ++blk) {
    float dl[8];
#pragma unroll
    for (int r = 0; r < 8; ++r) {
      const float* tp = t_in + (rowbase + r + hi8) * TGRID + blk;
      dl[r] = tp[1] - tp[0];
    }

#pragma clang loop unroll(disable)
    for (int step = 0; step < 2; ++step) {
      // stage y = x (f16, A-fragment order)
#pragma unroll
      for (int kt = 0; kt < 8; ++kt)
#pragma unroll
        for (int r = 0; r < 8; ++r)
          xs[(kt >> 1) * 512 + (r + hi8) * 16 + aoff + ((kt & 1) << 3)] =
              (_Float16)xv[kt * 8 + r];

      float acc[64];
#pragma unroll
      for (int i = 0; i < 64; ++i) acc[i] = 0.0f;

#pragma clang loop unroll(disable)
      for (int stg = 0; stg < 4; ++stg) {
        // ---------- MLP eval: facc = MLP(y), y in xs ----------
        v16h xA[4];
#pragma unroll
        for (int f = 0; f < 4; ++f)
          xA[f] = *(const v16h*)&xs[f * 512 + lane * 16];

        v8f facc[8];
#pragma unroll
        for (int j = 0; j < 8; ++j)
          facc[j] = (v8f){0.f, 0.f, 0.f, 0.f, 0.f, 0.f, 0.f, 0.f};

#pragma clang loop unroll(disable)
        for (int nc = 0; nc < 16; ++nc) {
          v8f h0 = (v8f){0.f, 0.f, 0.f, 0.f, 0.f, 0.f, 0.f, 0.f};
          v8f h1 = (v8f){0.f, 0.f, 0.f, 0.f, 0.f, 0.f, 0.f, 0.f};
#pragma unroll
          for (int f = 0; f < 4; ++f) {
            v16h bw0 = *(const v16h*)&w1f[(f * 32 + 2 * nc) * 512 + lane * 16];
            v16h bw1 = *(const v16h*)&w1f[(f * 32 + 2 * nc + 1) * 512 + lane * 16];
            h0 = __builtin_amdgcn_wmma_f32_16x16x32_f16(false, xA[f], false, bw0,
                                                        (short)0, h0, false, false);
            h1 = __builtin_amdgcn_wmma_f32_16x16x32_f16(false, xA[f], false, bw1,
                                                        (short)0, h1, false, false);
          }
          float bb0 = b1s[nc * 32 + col];
          float bb1 = b1s[nc * 32 + 16 + col];
          // tanh + scatter h (16x32 chunk) into A-fragment order (reuse frag0 of xs;
          // xA already lives in registers)
#pragma unroll
          for (int r = 0; r < 8; ++r) {
            xs[(r + hi8) * 16 + aoff]     = (_Float16)fast_tanh(h0[r] + bb0);
            xs[(r + hi8) * 16 + aoff + 8] = (_Float16)fast_tanh(h1[r] + bb1);
          }
          v16h hA = *(const v16h*)&xs[lane * 16];
#pragma unroll
          for (int j = 0; j < 8; ++j) {
            v16h bw = *(const v16h*)&w2f[(nc * 8 + j) * 512 + lane * 16];
            facc[j] = __builtin_amdgcn_wmma_f32_16x16x32_f16(false, hA, false, bw,
                                                             (short)0, facc[j],
                                                             false, false);
          }
        }
        // ---------- fuse: f = (facc + b2)*dl ; acc += w*f ; stage next y ----------
        float wN = (stg == 1 || stg == 2) ? 2.0f : 1.0f;
        float cN = (stg == 2) ? HSTEP : 0.5f * HSTEP;  // coef for NEXT stage's y
#pragma unroll
        for (int j = 0; j < 8; ++j)
#pragma unroll
          for (int r = 0; r < 8; ++r) {
            float fe = (facc[j][r] + b2v[j]) * dl[r];
            acc[j * 8 + r] += wN * fe;
            if (stg < 3)
              xs[(j >> 1) * 512 + (r + hi8) * 16 + aoff + ((j & 1) << 3)] =
                  (_Float16)(xv[j * 8 + r] + cN * fe);
          }
      }
#pragma unroll
      for (int i = 0; i < 64; ++i) xv[i] += (HSTEP / 6.0f) * acc[i];
    }

    // emit this block's state: out[(row*NBLOCK + blk)*K + k]
#pragma unroll
    for (int kt = 0; kt < 8; ++kt)
#pragma unroll
      for (int r = 0; r < 8; ++r)
        out[((rowbase + r + hi8) * NBLOCK + blk) * K_DIM + kt * 16 + col] =
            xv[kt * 8 + r];
  }
}

extern "C" void kernel_launch(void* const* d_in, const int* in_sizes, int n_in,
                              void* d_out, int out_size, void* d_ws, size_t ws_size,
                              hipStream_t stream) {
  (void)in_sizes; (void)n_in; (void)out_size; (void)d_ws; (void)ws_size;
  const float* s  = (const float*)d_in[0];
  const float* t  = (const float*)d_in[1];
  const float* W1 = (const float*)d_in[2];
  const float* b1 = (const float*)d_in[3];
  const float* W2 = (const float*)d_in[4];
  const float* b2 = (const float*)d_in[5];
  float* out = (float*)d_out;
  hipLaunchKernelGGL(ode_rk4_wmma, dim3(NWG), dim3(NTHREADS), 0, stream,
                     s, t, W1, b1, W2, b2, out);
}
